// WaveFieldAttention_73194832658570
// MI455X (gfx1250) — matrix operand order repaired and checked by reading, not compiled, not run on hardware
//
#include <hip/hip_runtime.h>
#include <hip/hip_bf16.h>

// ---------------------------------------------------------------------------
// WaveFieldAttention for MI455X (gfx1250): bf16 WMMA GEMMs + banded-WMMA
// causal wave convolution (exact-to-fp32 512-tap truncation).
// Shapes fixed by reference: B=8, N=4096, D=512, H=8, Dh=64, G=4096.
// N==G==MAX_SEQ  =>  bilinear interp degenerates to identity mapping
// (w_lo=1, w_hi=0, idx_lo=min(i,4094)).
// Round 2: branch-free conv inner loop via zero-padded field rows (left pad
// 512, right pad 32) and zero-padded LDS kernel table.
// ---------------------------------------------------------------------------

#define BB   8
#define NN   4096
#define DD   512
#define HH   8
#define DH   64
#define GG   4096
#define BN   (BB * NN)        // 32768
#define TT   512              // conv kernel truncation (exp(-a*511) < 2e-11)
#define LDF  (TT + GG + 32)   // padded field row stride (4640 halves)

typedef __attribute__((ext_vector_type(16))) __bf16 v16bf;
typedef __attribute__((ext_vector_type(8)))  __bf16 v8bf;
typedef __attribute__((ext_vector_type(8)))  float  v8f;

__device__ __forceinline__ v8f wmma_bf16(v16bf a, v16bf b, v8f c) {
    // D = A(16x32 bf16) * B(32x16 bf16) + C(16x16 f32)
    return __builtin_amdgcn_wmma_f32_16x16x32_bf16(
        /*neg_a=*/false, a, /*neg_b=*/false, b,
        /*c_mod=*/(short)0, c, /*reuse_a=*/false, /*reuse_b=*/false);
}

__device__ __forceinline__ v8f zero8() {
    v8f z;
#pragma unroll
    for (int i = 0; i < 8; ++i) z[i] = 0.0f;
    return z;
}

// A fragment (16x32 bf16) from row-major [rows, lda]: lane l holds row
// row0 + (l&15); halves 0-15 = contiguous K run at ((l>>4)<<4).
__device__ __forceinline__ v16bf load_afrag(const __bf16* __restrict__ A,
                                            int row0, int k0, int lda) {
    const int l = threadIdx.x & 31;
    const __bf16* p = A + (size_t)(row0 + (l & 15)) * lda + k0 + ((l >> 4) << 4);
    v8bf lo = *(const v8bf*)(p);
    v8bf hi = *(const v8bf*)(p + 8);
    v16bf r;
#pragma unroll
    for (int i = 0; i < 8; ++i) { r[i] = lo[i]; r[i + 8] = hi[i]; }
    return r;
}

// B fragment from pre-swizzled weights: 32 lanes * 16 halves contiguous.
__device__ __forceinline__ v16bf load_bfrag(const __bf16* __restrict__ Bsw,
                                            int nblk, int kblk, int kblks) {
    const int l = threadIdx.x & 31;
    return *(const v16bf*)(Bsw + ((((size_t)nblk * kblks + kblk) * 32 + l) << 4));
}

// ---------------------------------------------------------------------------
// Elementwise convert fp32 -> bf16
__global__ void k_cvt(const float* __restrict__ X, __bf16* __restrict__ Y, int n) {
    int i = blockIdx.x * 256 + threadIdx.x;
    if (i < n) Y[i] = (__bf16)X[i];
}

// Zero-fill a bf16 buffer
__global__ void k_zero_bf16(__bf16* __restrict__ p, int n) {
    int i = blockIdx.x * 256 + threadIdx.x;
    if (i < n) p[i] = (__bf16)0.0f;
}

// Swizzle weight W (K x Nw, row-major fp32) into B-fragment layout:
// SW[(nblk*kblks + kblk)*32 + lane][16 halves]; element (K,N):
// lane = (N&15) + ((K>>4 & 1)<<4), half = K&15 within a 32-K block.
__global__ void k_swizzle_w(const float* __restrict__ W, __bf16* __restrict__ SW,
                            int K, int Nw) {
    size_t i = (size_t)blockIdx.x * 256 + threadIdx.x;
    if (i >= (size_t)K * Nw) return;
    int h    = (int)(i & 15);
    int lane = (int)((i >> 4) & 31);
    size_t fi = i >> 9;
    int kblks = K >> 5;
    int kblk = (int)(fi % kblks);
    int nblk = (int)(fi / kblks);
    int kk = (kblk << 5) + ((lane >> 4) << 4) + h;
    int nn = (nblk << 4) + (lane & 15);
    SW[i] = (__bf16)W[(size_t)kk * Nw + nn];
}

// ---------------------------------------------------------------------------
// QKV GEMM: xb[BN,512] @ wqkv[512,1536] + b -> q,k,v as (B,H,N,Dh) fp32.
__global__ void __launch_bounds__(256) k_gemm_qkv(
    const __bf16* __restrict__ xb, const __bf16* __restrict__ wsw,
    const float* __restrict__ bqkv,
    float* __restrict__ q, float* __restrict__ kq, float* __restrict__ v) {
    const int l  = threadIdx.x & 31;
    const int wv = threadIdx.x >> 5;
    const int m0 = blockIdx.x * 32;
    const int n0 = blockIdx.y * 512 + wv * 64;
    const int nblk0 = n0 >> 4;
    v8f acc[2][4];
#pragma unroll
    for (int a = 0; a < 2; ++a)
#pragma unroll
        for (int b = 0; b < 4; ++b) acc[a][b] = zero8();

    for (int kb = 0; kb < 16; ++kb) {
        __builtin_prefetch(xb + (size_t)(m0 + (l & 15)) * 512 + kb * 32 + 32, 0, 1);
        v16bf a0 = load_afrag(xb, m0,      kb * 32, 512);
        v16bf a1 = load_afrag(xb, m0 + 16, kb * 32, 512);
#pragma unroll
        for (int nf = 0; nf < 4; ++nf) {
            v16bf bf = load_bfrag(wsw, nblk0 + nf, kb, 16);
            acc[0][nf] = wmma_bf16(a0, bf, acc[0][nf]);
            acc[1][nf] = wmma_bf16(a1, bf, acc[1][nf]);
        }
    }
#pragma unroll
    for (int nf = 0; nf < 4; ++nf) {
        int col  = n0 + nf * 16 + (l & 15);
        float bias = bqkv[col];
        int part = col >> 9;          // 0=q 1=k 2=v
        int rem  = col & 511;
        float* dst = (part == 0) ? q : ((part == 1) ? kq : v);
        size_t chan = (size_t)(rem >> 6) * (NN * DH) + (rem & 63);  // h*N*Dh + d
#pragma unroll
        for (int mt = 0; mt < 2; ++mt)
#pragma unroll
            for (int r = 0; r < 8; ++r) {
                int row = m0 + mt * 16 + r + ((l >> 4) << 3);
                int b = row >> 12, tn = row & 4095;
                dst[(size_t)b * (HH * NN * DH) + chan + (size_t)tn * DH] =
                    acc[mt][nf][r] + bias;
            }
    }
}

// Gate GEMM: sigmoid(xb @ wgate + b) -> bf16 [BN,512]
__global__ void __launch_bounds__(256) k_gemm_gate(
    const __bf16* __restrict__ xb, const __bf16* __restrict__ wsw,
    const float* __restrict__ bg, __bf16* __restrict__ gate) {
    const int l  = threadIdx.x & 31;
    const int wv = threadIdx.x >> 5;
    const int m0 = blockIdx.x * 32;
    const int n0 = wv * 64;
    v8f acc[2][4];
#pragma unroll
    for (int a = 0; a < 2; ++a)
#pragma unroll
        for (int b = 0; b < 4; ++b) acc[a][b] = zero8();
    for (int kb = 0; kb < 16; ++kb) {
        v16bf a0 = load_afrag(xb, m0,      kb * 32, 512);
        v16bf a1 = load_afrag(xb, m0 + 16, kb * 32, 512);
#pragma unroll
        for (int nf = 0; nf < 4; ++nf) {
            v16bf bf = load_bfrag(wsw, (n0 >> 4) + nf, kb, 16);
            acc[0][nf] = wmma_bf16(a0, bf, acc[0][nf]);
            acc[1][nf] = wmma_bf16(a1, bf, acc[1][nf]);
        }
    }
#pragma unroll
    for (int nf = 0; nf < 4; ++nf) {
        int col = n0 + nf * 16 + (l & 15);
        float bias = bg[col];
#pragma unroll
        for (int mt = 0; mt < 2; ++mt)
#pragma unroll
            for (int r = 0; r < 8; ++r) {
                int row = m0 + mt * 16 + r + ((l >> 4) << 3);
                float val = acc[mt][nf][r] + bias;
                gate[(size_t)row * DD + col] = (__bf16)(1.0f / (1.0f + expf(-val)));
            }
    }
}

// Output GEMM: ybf @ wout + b_out -> fp32 d_out [BN,512]
__global__ void __launch_bounds__(256) k_gemm_out(
    const __bf16* __restrict__ ybf, const __bf16* __restrict__ wsw,
    const float* __restrict__ bo, float* __restrict__ out) {
    const int l  = threadIdx.x & 31;
    const int wv = threadIdx.x >> 5;
    const int m0 = blockIdx.x * 32;
    const int n0 = wv * 64;
    v8f acc[2][4];
#pragma unroll
    for (int a = 0; a < 2; ++a)
#pragma unroll
        for (int b = 0; b < 4; ++b) acc[a][b] = zero8();
    for (int kb = 0; kb < 16; ++kb) {
        v16bf a0 = load_afrag(ybf, m0,      kb * 32, 512);
        v16bf a1 = load_afrag(ybf, m0 + 16, kb * 32, 512);
#pragma unroll
        for (int nf = 0; nf < 4; ++nf) {
            v16bf bf = load_bfrag(wsw, (n0 >> 4) + nf, kb, 16);
            acc[0][nf] = wmma_bf16(a0, bf, acc[0][nf]);
            acc[1][nf] = wmma_bf16(a1, bf, acc[1][nf]);
        }
    }
#pragma unroll
    for (int nf = 0; nf < 4; ++nf) {
        int col = n0 + nf * 16 + (l & 15);
        float bias = bo[col];
#pragma unroll
        for (int mt = 0; mt < 2; ++mt)
#pragma unroll
            for (int r = 0; r < 8; ++r) {
                int row = m0 + mt * 16 + r + ((l >> 4) << 3);
                out[(size_t)row * DD + col] = acc[mt][nf][r] + bias;
            }
    }
}

// ---------------------------------------------------------------------------
// ||k|| / sqrt(Dh) per token: one wave per (b,h,n)
__global__ void k_kmag(const float* __restrict__ kq, float* __restrict__ kmag) {
    int token = blockIdx.x * 8 + (threadIdx.x >> 5);   // b*H*N + h*N + n
    int l = threadIdx.x & 31;
    const float* p = kq + (size_t)token * DH;
    float a = p[l], b = p[l + 32];
    float s = a * a + b * b;
#pragma unroll
    for (int off = 16; off > 0; off >>= 1) s += __shfl_xor(s, off, 32);
    if (l == 0) kmag[token] = sqrtf(s) * 0.125f;       // /sqrt(64)
}

// field (transposed, padded, bf16): fieldT[bh][d][TT + g] = v[bh][g][d]*kmag
// (identity scatter; g==4094 also accumulates token 4095; g==4095 stays 0;
// pad regions pre-zeroed by k_zero_bf16).
__global__ void k_field(const float* __restrict__ v, const float* __restrict__ kmag,
                        __bf16* __restrict__ fieldT) {
    int idx = blockIdx.x * 8 + (threadIdx.x >> 5);     // bh*G + g
    int g = idx & (GG - 1);
    int bh = idx >> 12;
    int l = threadIdx.x & 31;
    float d0 = 0.0f, d1 = 0.0f;
    if (g < GG - 1) {
        size_t base = ((size_t)bh * NN + g) * DH;
        float km = kmag[bh * NN + g];
        d0 = v[base + l] * km;
        d1 = v[base + l + 32] * km;
        if (g == GG - 2) {
            float km2 = kmag[bh * NN + (NN - 1)];
            d0 += v[base + DH + l] * km2;
            d1 += v[base + DH + l + 32] * km2;
        }
    }
    __bf16* fp = fieldT + (size_t)bh * (DH * LDF) + TT + g;
    fp[(size_t)l * LDF]        = (__bf16)d0;
    fp[(size_t)(l + 32) * LDF] = (__bf16)d1;
}

// Damped-wave kernel table: kern[h][t] = exp(-softplus(damp)*t)*cos(f*t+phi)
__global__ void k_wavekern(const float* __restrict__ freq, const float* __restrict__ damp,
                           const float* __restrict__ phase, __bf16* __restrict__ kern) {
    int i = blockIdx.x * 256 + threadIdx.x;
    if (i >= HH * TT) return;
    int h = i >> 9, t = i & (TT - 1);
    float a = log1pf(expf(damp[h]));
    float ft = (float)t;
    kern[i] = (__bf16)(expf(-a * ft) * cosf(freq[h] * ft + phase[h]));
}

// Banded causal convolution via WMMA:
// convT[bh][d][g] = sum_{t=0..511} kern[h][t] * fieldT[bh][d][TT + g - t]
// Wave tile: M=64 d-rows (4 A frags) x N=16 g-cols; K loop over 17 chunks of
// 32 source positions. Branch-free: field rows are zero-padded (left TT,
// right 32), and the LDS kernel table has 32 zero entries on each side, so
// no bounds checks are needed anywhere in the hot loop.
__global__ void __launch_bounds__(128) k_conv(const __bf16* __restrict__ fieldT,
                                              const __bf16* __restrict__ kern,
                                              float* __restrict__ convT) {
    __shared__ __bf16 skp[TT + 64];                    // [32 zeros | kern | 32 zeros]
    const int bh = blockIdx.x;
    const int h  = bh & (HH - 1);
    for (int i = threadIdx.x; i < TT + 64; i += 128) {
        int t = i - 32;
        skp[i] = (t >= 0 && t < TT) ? kern[h * TT + t] : (__bf16)0.0f;
    }
    __syncthreads();

    const int l  = threadIdx.x & 31;
    const int wv = threadIdx.x >> 5;
    const int g0 = blockIdx.y * 64 + wv * 16;
    const __bf16* F = fieldT + (size_t)bh * (DH * LDF);
    const int g = g0 + (l & 15);

    v8f acc[4];
#pragma unroll
    for (int i = 0; i < 4; ++i) acc[i] = zero8();

    for (int cc = 0; cc < 17; ++cc) {
        const int sb = g0 - TT + cc * 32;              // source-chunk base
        // Band fragment: B[K][N] = kern[g - s]; padded LDS -> no compares.
        const int tb = g - (sb + ((l >> 4) << 4)) + 32;  // in [1, 559]
        v16bf bb;
#pragma unroll
        for (int i = 0; i < 16; ++i) bb[i] = skp[tb - i];
        // A fragments: padded rows -> unconditional aligned 16B runs.
        const int s0 = TT + sb + ((l >> 4) << 4);        // >= 0 by construction
#pragma unroll
        for (int mf = 0; mf < 4; ++mf) {
            const __bf16* prow = F + (size_t)(mf * 16 + (l & 15)) * LDF + s0;
            v8bf t0 = *(const v8bf*)(prow);
            v8bf t1 = *(const v8bf*)(prow + 8);
            v16bf aa;
#pragma unroll
            for (int i = 0; i < 8; ++i) { aa[i] = t0[i]; aa[i + 8] = t1[i]; }
            acc[mf] = wmma_bf16(aa, bb, acc[mf]);
        }
    }
    float* O = convT + (size_t)bh * (DH * GG) + g0 + (l & 15);
#pragma unroll
    for (int mf = 0; mf < 4; ++mf)
#pragma unroll
        for (int r = 0; r < 8; ++r)
            O[(size_t)(mf * 16 + r + ((l >> 4) << 3)) * GG] = acc[mf][r];
}

// Coupling (hk) + identity gather (g=min(n,4094)) + sigmoid(q/8) + gate -> bf16 y
__global__ void k_modulate(const float* __restrict__ convT, const float* __restrict__ q,
                           const __bf16* __restrict__ gate, const float* __restrict__ coupling,
                           __bf16* __restrict__ ybf) {
    __shared__ float sc[HH * HH];
    if (threadIdx.x < HH * HH) sc[threadIdx.x] = coupling[threadIdx.x];
    __syncthreads();
    int idx = blockIdx.x * 8 + (threadIdx.x >> 5);     // b*N + n
    int l = threadIdx.x & 31;
    int b = idx >> 12, n = idx & (NN - 1);
    int g = (n < GG - 2) ? n : (GG - 2);
    float cv0[HH], cv1[HH];
#pragma unroll
    for (int h = 0; h < HH; ++h) {
        const float* p = convT + ((size_t)(b * HH + h) * DH) * GG + g;
        cv0[h] = p[(size_t)l * GG];
        cv1[h] = p[(size_t)(l + 32) * GG];
    }
#pragma unroll
    for (int kh = 0; kh < HH; ++kh) {
        float s0 = 0.0f, s1 = 0.0f;
#pragma unroll
        for (int h = 0; h < HH; ++h) {
            float c = sc[h * HH + kh];
            s0 += c * cv0[h];
            s1 += c * cv1[h];
        }
        size_t qb = ((size_t)(b * HH + kh) * NN + n) * DH;
        float m0 = 1.0f / (1.0f + expf(-q[qb + l] * 0.125f));
        float m1 = 1.0f / (1.0f + expf(-q[qb + l + 32] * 0.125f));
        size_t yb = (size_t)idx * DD + kh * DH;
        float g0 = (float)gate[yb + l];
        float g1 = (float)gate[yb + l + 32];
        ybf[yb + l]      = (__bf16)(s0 * m0 * g0);
        ybf[yb + l + 32] = (__bf16)(s1 * m1 * g1);
    }
}

// ---------------------------------------------------------------------------
extern "C" void kernel_launch(void* const* d_in, const int* in_sizes, int n_in,
                              void* d_out, int out_size, void* d_ws, size_t ws_size,
                              hipStream_t stream) {
    const float* x        = (const float*)d_in[0];
    const float* w_qkv    = (const float*)d_in[1];
    const float* b_qkv    = (const float*)d_in[2];
    const float* w_gate   = (const float*)d_in[3];
    const float* b_gate   = (const float*)d_in[4];
    const float* w_out    = (const float*)d_in[5];
    const float* b_out    = (const float*)d_in[6];
    const float* wfreq    = (const float*)d_in[7];
    const float* wdamp    = (const float*)d_in[8];
    const float* wphase   = (const float*)d_in[9];
    const float* coupling = (const float*)d_in[10];
    float* out = (float*)d_out;

    size_t off = 0;
    auto wsAlloc = [&](size_t bytes) {
        char* p = (char*)d_ws + off;
        off += (bytes + 255) & ~(size_t)255;
        return (void*)p;
    };
    __bf16* xb       = (__bf16*)wsAlloc((size_t)BN * DD * 2);
    __bf16* wqkv_sw  = (__bf16*)wsAlloc((size_t)DD * 3 * DD * 2);
    __bf16* wgate_sw = (__bf16*)wsAlloc((size_t)DD * DD * 2);
    __bf16* wout_sw  = (__bf16*)wsAlloc((size_t)DD * DD * 2);
    float*  qbuf     = (float*) wsAlloc((size_t)BN * DD * 4);
    float*  kbuf     = (float*) wsAlloc((size_t)BN * DD * 4);   // 67MB, > padded field (38MB)
    float*  vbuf     = (float*) wsAlloc((size_t)BN * DD * 4);
    float*  kmag     = (float*) wsAlloc((size_t)BB * HH * NN * 4);
    __bf16* kern     = (__bf16*)wsAlloc((size_t)HH * TT * 2);
    __bf16* gate     = (__bf16*)wsAlloc((size_t)BN * DD * 2);
    __bf16* ybf      = (__bf16*)wsAlloc((size_t)BN * DD * 2);
    // Aliases (lifetimes disjoint): field reuses k's space, conv reuses v's.
    __bf16* fieldT = (__bf16*)kbuf;            // (B*H, Dh, LDF) bf16, padded
    float*  convT  = vbuf;                     // (B*H, Dh, G) fp32

    const int nfield = BB * HH * DH * LDF;     // 19,005,440 halves

    // 1. Converts / swizzles / kernel table
    k_cvt<<<(BN * DD + 255) / 256, 256, 0, stream>>>(x, xb, BN * DD);
    k_swizzle_w<<<(DD * 3 * DD + 255) / 256, 256, 0, stream>>>(w_qkv, wqkv_sw, DD, 3 * DD);
    k_swizzle_w<<<(DD * DD + 255) / 256, 256, 0, stream>>>(w_gate, wgate_sw, DD, DD);
    k_swizzle_w<<<(DD * DD + 255) / 256, 256, 0, stream>>>(w_out, wout_sw, DD, DD);
    k_wavekern<<<(HH * TT + 255) / 256, 256, 0, stream>>>(wfreq, wdamp, wphase, kern);

    // 2. QKV projection (WMMA)
    k_gemm_qkv<<<dim3(BN / 32, 3), 256, 0, stream>>>(xb, wqkv_sw, b_qkv, qbuf, kbuf, vbuf);

    // 3. Deposit field (zero pads first; k_field fills the interior)
    k_kmag <<<BB * HH * NN / 8, 256, 0, stream>>>(kbuf, kmag);
    k_zero_bf16<<<(nfield + 255) / 256, 256, 0, stream>>>(fieldT, nfield);
    k_field<<<BB * HH * GG / 8, 256, 0, stream>>>(vbuf, kmag, fieldT);

    // 4. Banded causal wave convolution (WMMA, branch-free hot loop)
    k_conv<<<dim3(BB * HH, GG / 64), 128, 0, stream>>>(fieldT, kern, convT);

    // 5. Content gate (WMMA) + coupling/gather/modulation
    k_gemm_gate<<<dim3(BN / 32, 1), 256, 0, stream>>>(xb, wgate_sw, b_gate, gate);
    k_modulate <<<BB * NN / 8, 256, 0, stream>>>(convT, qbuf, gate, coupling, ybf);

    // 6. Output projection (WMMA)
    k_gemm_out<<<dim3(BN / 32, 1), 256, 0, stream>>>(ybf, wout_sw, b_out, out);

    (void)in_sizes; (void)n_in; (void)out_size; (void)ws_size;
}
